// AdaptiveReluMPNN_85624468013530
// MI455X (gfx1250) — matrix operand order repaired and checked by reading, not compiled, not run on hardware
//
#include <hip/hip_runtime.h>
#include <hip/hip_bf16.h>
#include <stdint.h>

// Problem constants (match reference)
#define NN   100000
#define EE   1600000
#define FF   64
#define GG   128
#define OUTD 32
#define LLAY 2

typedef __attribute__((ext_vector_type(16))) __bf16          v16bf;
typedef __attribute__((ext_vector_type(16))) unsigned short  v16us;
typedef __attribute__((ext_vector_type(8)))  float           v8f;

// ---------- helpers ----------
__device__ __forceinline__ unsigned short f2bf(float f) {
  // fp32 -> bf16 bits, round-to-nearest-even
  unsigned u = __float_as_uint(f);
  u += 0x7FFFu + ((u >> 16) & 1u);
  return (unsigned short)(u >> 16);
}
// monotone bijection float <-> uint so uint atomicMin/Max == float min/max
__device__ __forceinline__ unsigned fkey(float f) {
  unsigned u = __float_as_uint(f);
  return (u & 0x80000000u) ? ~u : (u | 0x80000000u);
}
__device__ __forceinline__ float funkey(unsigned k) {
  unsigned u = (k & 0x80000000u) ? (k & 0x7FFFFFFFu) : ~k;
  return __uint_as_float(u);
}

// Shared WMMA micro-GEMM: 32x64 output tile per 256-thread block (8 waves,
// one 16x16 tile each), K=64 as two 16x16x32 bf16 WMMAs.
// As[r][k]   : 32 rows of A (bf16 bits), k contiguous  -> 2x b128 per frag
// Bs[n][k]   : 64 cols of B^T == W native [out][in]    -> 2x b128 per frag
__device__ __forceinline__ v8f wmma_tile_32x64(
    const unsigned short (*__restrict__ As)[FF],
    const unsigned short (*__restrict__ Bs)[FF],
    int rowA, int colN, int aoff, int boff)
{
  v8f acc = {};
#pragma unroll
  for (int ks = 0; ks < 2; ++ks) {
    const int kb = ks * 32;
    v16us au, bu;
#pragma unroll
    for (int i = 0; i < 8; ++i) {
      au[i]     = As[rowA][kb + aoff + i];        // contiguous 16B run
      au[i + 8] = As[rowA][kb + aoff + 8 + i];    // contiguous 16B run
    }
#pragma unroll
    for (int i = 0; i < 16; ++i) bu[i] = Bs[colN][kb + boff + i];  // contiguous 32B
    v16bf a = __builtin_bit_cast(v16bf, au);
    v16bf b = __builtin_bit_cast(v16bf, bu);
    acc = __builtin_amdgcn_wmma_f32_16x16x32_bf16(false, a, false, b, (short)0, acc,
                                                  false, false);
  }
  return acc;
}

// ---------- tiny utility kernels ----------
__global__ void zero_int_kernel(int* __restrict__ p, int n) {
  int i = blockIdx.x * 256 + threadIdx.x;
  if (i < n) p[i] = 0;
}
__global__ void count_kernel(const int* __restrict__ idx, int* __restrict__ cnt, int n) {
  int i = blockIdx.x * 256 + threadIdx.x;
  if (i < n) atomicAdd(&cnt[idx[i]], 1);
}
__global__ void init_stats_kernel(unsigned* __restrict__ mn, unsigned* __restrict__ mx,
                                  float* __restrict__ sm, float* __restrict__ rl, int n) {
  int i = blockIdx.x * 256 + threadIdx.x;
  if (i < n) { mn[i] = 0xFFFFFFFFu; mx[i] = 0u; sm[i] = 0.0f; rl[i] = 0.0f; }
}

// ---------- edge message GEMM + segment stats (WMMA bf16) ----------
// Block: 256 threads (8 waves). Tile: 32 edges x 64 features.
__global__ __launch_bounds__(256)
void edge_msg_kernel(const float* __restrict__ h, const int* __restrict__ src,
                     const int* __restrict__ dst, const float* __restrict__ W,
                     const float* __restrict__ bias, const float* __restrict__ tvec,
                     unsigned* __restrict__ mnk, unsigned* __restrict__ mxk,
                     float* __restrict__ smb, float* __restrict__ rlb, int pass)
{
  __shared__ unsigned short As[32][FF];   // gathered h rows, bf16 bits
  __shared__ unsigned short Bs[FF][FF];   // Bs[n][k] = bf16(W[n][k]) (native W layout)
  __shared__ int dsts[32];

  const int  tid   = threadIdx.x;
  const long ebase = (long)blockIdx.x * 32;

  for (int i = tid; i < FF * FF; i += 256)           // straight copy, no transpose
    (&Bs[0][0])[i] = f2bf(W[i]);
  if (tid < 32) dsts[tid] = dst[ebase + tid];
  for (int i = tid; i < 32 * FF; i += 256) {
    int r = i >> 6, c = i & 63;
    int node = src[ebase + r];
    As[r][c] = f2bf(h[(size_t)node * FF + c]);
  }
  __syncthreads();

  const int lane = tid & 31;
  const int wave = tid >> 5;
  const int mt   = wave >> 2;                 // 0..1
  const int nt   = wave & 3;                  // 0..3
  const int rowA = mt * 16 + (lane & 15);     // edge row within tile
  const int colN = nt * 16 + (lane & 15);     // feature column 0..63
  const int aoff = (lane & 16) ? 16 : 0;      // A: hi lanes hold K 16..31
  const int boff = (lane & 16) ? 16 : 0;      // B: hi lanes hold K 16..31

  v8f acc = wmma_tile_32x64(As, Bs, rowA, colN, aoff, boff);

  // C layout: vgpr r -> M = r + (lane>=16 ? 8 : 0); N = lane&15
  const int   mloc = mt * 16 + ((lane & 16) ? 8 : 0);
  const float bn   = bias[colN];
  if (pass == 0) {
#pragma unroll
    for (int r = 0; r < 8; ++r) {
      float  v    = acc[r] + bn;
      int    d    = dsts[mloc + r];
      size_t sidx = (size_t)d * FF + colN;
      atomicMin(&mnk[sidx], fkey(v));
      atomicMax(&mxk[sidx], fkey(v));
      atomicAdd(&smb[sidx], v);
    }
  } else {
    const float tn = fminf(fmaxf(tvec[colN], 0.0f), 1.0f);
#pragma unroll
    for (int r = 0; r < 8; ++r) {
      float  v    = acc[r] + bn;
      int    d    = dsts[mloc + r];
      size_t sidx = (size_t)d * FF + colN;
      float  mnv  = funkey(mnk[sidx]);
      float  mxv  = funkey(mxk[sidx]);
      float  bb   = tn * mxv + (1.0f - tn) * mnv;
      atomicAdd(&rlb[sidx], fmaxf(v - bb, 0.0f));
    }
  }
}

// ---------- node GEMM (WMMA bf16): out = h @ W^T + bias [+ agg(stats)] ----------
__global__ __launch_bounds__(256)
void node_gemm_kernel(const float* __restrict__ h, const float* __restrict__ W,
                      const float* __restrict__ bias, float* __restrict__ out,
                      const unsigned* __restrict__ mnk, const unsigned* __restrict__ mxk,
                      const float* __restrict__ smb, const float* __restrict__ rlb,
                      const int* __restrict__ cnt, const float* __restrict__ Wp,
                      const float* __restrict__ bp, int addAgg)
{
  __shared__ unsigned short As[32][FF];
  __shared__ unsigned short Bs[FF][FF];

  const int  tid   = threadIdx.x;
  const long rbase = (long)blockIdx.x * 32;

  for (int i = tid; i < FF * FF; i += 256)
    (&Bs[0][0])[i] = f2bf(W[i]);
  for (int i = tid; i < 32 * FF; i += 256) {
    int r = i >> 6, c = i & 63;
    As[r][c] = f2bf(h[(size_t)(rbase + r) * FF + c]);
  }
  __syncthreads();

  const int lane = tid & 31;
  const int wave = tid >> 5;
  const int mt   = wave >> 2;
  const int nt   = wave & 3;
  const int rowA = mt * 16 + (lane & 15);
  const int colN = nt * 16 + (lane & 15);
  const int aoff = (lane & 16) ? 16 : 0;
  const int boff = (lane & 16) ? 16 : 0;

  v8f acc = wmma_tile_32x64(As, Bs, rowA, colN, aoff, boff);

  const int   mloc = mt * 16 + ((lane & 16) ? 8 : 0);
  const float bn   = bias[colN];
#pragma unroll
  for (int r = 0; r < 8; ++r) {
    long  row = rbase + mloc + r;
    float v   = acc[r] + bn;
    if (addAgg) {
      size_t sidx = (size_t)row * FF + colN;
      float  mnv  = funkey(mnk[sidx]);
      float  mxv  = funkey(mxk[sidx]);
      v += Wp[0] * (float)cnt[row] + Wp[1] * mnv + Wp[2] * mxv +
           Wp[3] * rlb[sidx] + Wp[4] * smb[sidx] + bp[0];
    }
    out[(size_t)row * FF + colN] = v;
  }
}

// ---------- graph-level segment stats over gmsg [N,64] -> [G,64] ----------
__global__ __launch_bounds__(256)
void graph_pass_kernel(const float* __restrict__ gmsg, const int* __restrict__ batch,
                       unsigned* __restrict__ mnk, unsigned* __restrict__ mxk,
                       float* __restrict__ smb, float* __restrict__ rlb,
                       const float* __restrict__ gt, int pass)
{
  int i = blockIdx.x * 256 + threadIdx.x;
  if (i >= NN * FF) return;
  int    row  = i >> 6;
  int    f    = i & 63;
  int    g    = batch[row];
  float  v    = gmsg[i];
  size_t sidx = (size_t)g * FF + f;
  if (pass == 0) {
    atomicMin(&mnk[sidx], fkey(v));
    atomicMax(&mxk[sidx], fkey(v));
    atomicAdd(&smb[sidx], v);
  } else {
    float tn = fminf(fmaxf(gt[f], 0.0f), 1.0f);
    float bb = tn * funkey(mxk[sidx]) + (1.0f - tn) * funkey(mnk[sidx]);
    atomicAdd(&rlb[sidx], fmaxf(v - bb, 0.0f));
  }
}

// ---------- final: emb = coords . gWproj + gbproj ; out = emb @ Wout^T + bout ----------
__global__ __launch_bounds__(64)
void final_kernel(const unsigned* __restrict__ mnk, const unsigned* __restrict__ mxk,
                  const float* __restrict__ smb, const float* __restrict__ rlb,
                  const int* __restrict__ gcnt, const float* __restrict__ gWproj,
                  const float* __restrict__ gbproj, const float* __restrict__ Wout,
                  const float* __restrict__ bout, float* __restrict__ out)
{
  __shared__ float emb[FF];
  const int g = blockIdx.x;
  const int f = threadIdx.x;                 // 64 threads
  size_t sidx = (size_t)g * FF + f;
  float  mnv  = funkey(mnk[sidx]);
  float  mxv  = funkey(mxk[sidx]);
  float  cf   = (float)gcnt[g];
  emb[f] = gWproj[0] * cf + gWproj[1] * mnv + gWproj[2] * mxv +
           gWproj[3] * rlb[sidx] + gWproj[4] * smb[sidx] + gbproj[0];
  __syncthreads();
  if (f < OUTD) {
    float acc = bout[f];
#pragma unroll
    for (int k = 0; k < FF; ++k) acc += emb[k] * Wout[f * FF + k];
    out[(size_t)g * OUTD + f] = acc;
  }
}

// ---------- launch ----------
extern "C" void kernel_launch(void* const* d_in, const int* in_sizes, int n_in,
                              void* d_out, int out_size, void* d_ws, size_t ws_size,
                              hipStream_t stream) {
  const float* x      = (const float*)d_in[0];
  const int*   eidx   = (const int*)d_in[1];
  const int*   batch  = (const int*)d_in[2];
  const float* Wlin   = (const float*)d_in[3];
  const float* blin   = (const float*)d_in[4];
  const float* tpar   = (const float*)d_in[5];
  const float* Wproj  = (const float*)d_in[6];
  const float* bproj  = (const float*)d_in[7];
  const float* Wc     = (const float*)d_in[8];
  const float* bc     = (const float*)d_in[9];
  const float* gWlin  = (const float*)d_in[10];
  const float* gblin  = (const float*)d_in[11];
  const float* gt     = (const float*)d_in[12];
  const float* gWproj = (const float*)d_in[13];
  const float* gbproj = (const float*)d_in[14];
  const float* Wout   = (const float*)d_in[15];
  const float* bout   = (const float*)d_in[16];
  const int*   src    = eidx;
  const int*   dstp   = eidx + EE;

  char*  ws  = (char*)d_ws;
  size_t off = 0;
  auto alloc = [&](size_t bytes) -> void* {
    void* p = ws + off;
    off += bytes;
    off = (off + 255) & ~(size_t)255;
    return p;
  };
  float*    h0   = (float*)alloc((size_t)NN * FF * 4);
  float*    h1   = (float*)alloc((size_t)NN * FF * 4);
  unsigned* mnk  = (unsigned*)alloc((size_t)NN * FF * 4);
  unsigned* mxk  = (unsigned*)alloc((size_t)NN * FF * 4);
  float*    smb  = (float*)alloc((size_t)NN * FF * 4);
  float*    rlb  = (float*)alloc((size_t)NN * FF * 4);
  int*      cnt  = (int*)alloc((size_t)NN * 4);
  int*      gcnt = (int*)alloc((size_t)GG * 4);
  unsigned* gmn  = (unsigned*)alloc((size_t)GG * FF * 4);
  unsigned* gmx  = (unsigned*)alloc((size_t)GG * FF * 4);
  float*    gsm  = (float*)alloc((size_t)GG * FF * 4);
  float*    grl  = (float*)alloc((size_t)GG * FF * 4);
  float*    gmsg = h0;  // h0 is free once layer 1 combine has written h1

  // in-degree per node (same both layers) + nodes per graph
  zero_int_kernel<<<(NN + 255) / 256, 256, 0, stream>>>(cnt, NN);
  zero_int_kernel<<<1, 256, 0, stream>>>(gcnt, GG);
  count_kernel<<<(EE + 255) / 256, 256, 0, stream>>>(dstp, cnt, EE);
  count_kernel<<<(NN + 255) / 256, 256, 0, stream>>>(batch, gcnt, NN);

  const float* hcur = x;
  float* hbufs[2] = {h0, h1};
  for (int l = 0; l < LLAY; ++l) {
    init_stats_kernel<<<(NN * FF + 255) / 256, 256, 0, stream>>>(mnk, mxk, smb, rlb, NN * FF);
    edge_msg_kernel<<<EE / 32, 256, 0, stream>>>(hcur, src, dstp,
        Wlin + (size_t)l * FF * FF, blin + l * FF, nullptr,
        mnk, mxk, smb, rlb, 0);
    edge_msg_kernel<<<EE / 32, 256, 0, stream>>>(hcur, src, dstp,
        Wlin + (size_t)l * FF * FF, blin + l * FF, tpar + l * FF,
        mnk, mxk, smb, rlb, 1);
    node_gemm_kernel<<<NN / 32, 256, 0, stream>>>(hcur,
        Wc + (size_t)l * FF * FF, bc + l * FF, hbufs[l],
        mnk, mxk, smb, rlb, cnt, Wproj + l * 5, bproj + l, 1);
    hcur = hbufs[l];
  }

  // global conv message: gmsg = h @ gWlin^T + gblin
  node_gemm_kernel<<<NN / 32, 256, 0, stream>>>(hcur, gWlin, gblin, gmsg,
      nullptr, nullptr, nullptr, nullptr, nullptr, nullptr, nullptr, 0);

  init_stats_kernel<<<(GG * FF + 255) / 256, 256, 0, stream>>>(gmn, gmx, gsm, grl, GG * FF);
  graph_pass_kernel<<<(NN * FF + 255) / 256, 256, 0, stream>>>(gmsg, batch, gmn, gmx, gsm, grl, gt, 0);
  graph_pass_kernel<<<(NN * FF + 255) / 256, 256, 0, stream>>>(gmsg, batch, gmn, gmx, gsm, grl, gt, 1);
  final_kernel<<<GG, 64, 0, stream>>>(gmn, gmx, gsm, grl, gcnt, gWproj, gbproj,
                                      Wout, bout, (float*)d_out);
}
